// ProjViewTransformer_46505905881646
// MI455X (gfx1250) — matrix operand order repaired and checked by reading, not compiled, not run on hardware
//
#include <hip/hip_runtime.h>
#include <hip/hip_bf16.h>

typedef __attribute__((ext_vector_type(16))) __bf16 v16bf;
typedef __attribute__((ext_vector_type(4)))  __bf16 v4bf;
typedef __attribute__((ext_vector_type(8)))  float  v8f;

#define B_    2
#define NC_   6
#define CH_   256
#define HF_   32
#define WF_   88
#define N_    100000
#define CIN_  128
#define HFWF_ (HF_ * WF_)
#define VPB   64      // voxels per block
#define TPB   128     // threads per block (4 wave32)

// ---------------------------------------------------------------------------
// Prep kernel: binv[b] = inv(bda[b,:3,:3]); L[b,c] = intrins[b,c] @ R[b,c]^T
// (reference's L[...,:3,3] is provably zero, so only the 3x3 part is needed)
// ws layout: [0..18) binv (B*9 floats), [18..126) L (B*NC*9 floats)
// ---------------------------------------------------------------------------
__global__ void proj_prep_kernel(const float* __restrict__ intrins,
                                 const float* __restrict__ bda,
                                 const float* __restrict__ l2c,
                                 float* __restrict__ ws)
{
    int t = threadIdx.x;
    if (t < B_) {
        const float* A = bda + t * 16;
        float a = A[0], b = A[1], c = A[2];
        float d = A[4], e = A[5], f = A[6];
        float g = A[8], h = A[9], i = A[10];
        float det = a * (e * i - f * h) - b * (d * i - f * g) + c * (d * h - e * g);
        float id = 1.0f / det;
        float* o = ws + t * 9;
        o[0] = (e * i - f * h) * id; o[1] = (c * h - b * i) * id; o[2] = (b * f - c * e) * id;
        o[3] = (f * g - d * i) * id; o[4] = (a * i - c * g) * id; o[5] = (c * d - a * f) * id;
        o[6] = (d * h - e * g) * id; o[7] = (b * g - a * h) * id; o[8] = (a * e - b * d) * id;
    }
    if (t < B_ * NC_) {
        const float* K = intrins + t * 9;
        const float* R = l2c + t * 16;   // rotation rows at [j*4 + k]
        float* o = ws + B_ * 9 + t * 9;
        #pragma unroll
        for (int ii = 0; ii < 3; ++ii)
            #pragma unroll
            for (int jj = 0; jj < 3; ++jj)
                o[ii * 3 + jj] = K[ii * 3 + 0] * R[jj * 4 + 0]
                               + K[ii * 3 + 1] * R[jj * 4 + 1]
                               + K[ii * 3 + 2] * R[jj * 4 + 2];
    }
}

// ---------------------------------------------------------------------------
// Main kernel: projection -> gather/sum -> bf16 WMMA GEMM (gsum @ W^T)
// FULL=true: whole block in-range (no bounds checks anywhere).
// ---------------------------------------------------------------------------
template <bool FULL>
__global__ void __launch_bounds__(TPB)
proj_view_kernel(const float* __restrict__ voxel_features,
                 const int*   __restrict__ voxel_coords,
                 const float* __restrict__ img_feats,
                 const float* __restrict__ post_rots,
                 const float* __restrict__ post_trans,
                 const float* __restrict__ bda,
                 const float* __restrict__ Wg,
                 const float* __restrict__ ws,
                 float*       __restrict__ out_feats,
                 int*         __restrict__ out_coords,
                 int block_base)
{
    extern __shared__ char smem[];
    __bf16* sW   = (__bf16*)smem;                        // [CIN_][CH_] bf16: 65536 B
    __bf16* sG   = (__bf16*)(smem + 65536);              // [4][16][CH_] bf16: 32768 B
    int*    sOff = (int*)(smem + 65536 + 32768);         // [VPB][NC_]: 1536 B

    const int tid  = threadIdx.x;
    const int lane = tid & 31;
    const int wid  = tid >> 5;
    const int base = (block_base + blockIdx.x) * VPB;

    // ---- Stage W (f32 -> bf16) into LDS, packed b64 stores -------------
    {
        const float4* W4 = (const float4*)Wg;
        for (int i = tid; i < CIN_ * CH_ / 4; i += TPB) {
            float4 w = W4[i];
            v4bf p = { (__bf16)w.x, (__bf16)w.y, (__bf16)w.z, (__bf16)w.w };
            *(v4bf*)(sW + i * 4) = p;
        }
    }

    // ---- Phase 1: projection (2 threads/voxel, 3 cams each) ------------
    {
        const int v  = tid >> 1;
        const int c0 = (tid & 1) * 3;
        const int n  = base + v;
        if (FULL || n < N_) {
            int4 c4 = ((const int4*)voxel_coords)[n];    // (b, z, y, x)
            const int b = c4.x;
            const float px = (float)c4.w * 0.075f + (-54.0f);
            const float py = (float)c4.z * 0.075f + (-54.0f);
            const float pz = (float)c4.y * 0.2f   + (-5.0f);
            const float* bdab = bda + b * 16;
            const float qx = px - bdab[3];
            const float qy = py - bdab[7];
            const float qz = pz - bdab[11];
            const float* bi = ws + b * 9;
            const float cx = bi[0] * qx + bi[1] * qy + bi[2] * qz;
            const float cy = bi[3] * qx + bi[4] * qy + bi[5] * qz;
            const float cz = bi[6] * qx + bi[7] * qy + bi[8] * qz;
            #pragma unroll
            for (int cc = 0; cc < 3; ++cc) {
                const int cam = c0 + cc;
                const int bc = b * NC_ + cam;
                const float* L = ws + B_ * 9 + bc * 9;
                const float ix = L[0] * cx + L[1] * cy + L[2] * cz;
                const float iy = L[3] * cx + L[4] * cy + L[5] * cz;
                const float iz = L[6] * cx + L[7] * cy + L[8] * cz;
                const float u = ix / iz, v2 = iy / iz, w = iz;
                const float* pr = post_rots + bc * 9;
                const float* pt = post_trans + bc * 3;
                const float sx = pr[0] * u + pr[1] * v2 + pr[2] * w + pt[0];
                const float sy = pr[3] * u + pr[4] * v2 + pr[5] * w + pt[1];
                const float sd = pr[6] * u + pr[7] * v2 + pr[8] * w + pt[2];
                const float cxr = rintf(sx * 0.125f);
                const float cyr = rintf(sy * 0.125f);
                int off = -1;
                if (cxr >= 0.0f && cxr < (float)WF_ &&
                    cyr >= 0.0f && cyr < (float)HF_ &&
                    sd < 60.0f && sd >= 1.0f) {
                    int xi = (int)cxr; xi = xi < 0 ? 0 : (xi > WF_ - 1 ? WF_ - 1 : xi);
                    int yi = (int)cyr; yi = yi < 0 ? 0 : (yi > HF_ - 1 ? HF_ - 1 : yi);
                    off = (bc * CH_ * HF_ + yi) * WF_ + xi;   // ch stride = HFWF_
                }
                sOff[v * NC_ + cam] = off;
            }
        } else {
            #pragma unroll
            for (int cc = 0; cc < 3; ++cc) sOff[v * NC_ + c0 + cc] = -1;
        }
    }

    // ---- Passthrough copies (global-only; overlaps with barrier) -------
    {
        const float4* vf = (const float4*)voxel_features;
        float4* of = (float4*)out_feats;
        for (int i = tid; i < VPB * (CIN_ / 4); i += TPB) {
            const int v = i >> 5, c4i = i & 31;
            const int n = base + v;
            if (FULL || n < N_)
                of[(size_t)n * (CIN_ / 4) + c4i] = vf[(size_t)n * (CIN_ / 4) + c4i];
        }
        const int4* vc = (const int4*)voxel_coords;
        int4* oc = (int4*)out_coords;
        for (int i = tid; i < VPB; i += TPB) {
            const int n = base + i;
            if (FULL || n < N_) { int4 c = vc[n]; oc[n] = c; oc[N_ + n] = c; }
        }
    }

    __syncthreads();

    // ---- Phase 2: masked gather + camera sum -> bf16 tile in LDS -------
    {
        __bf16* sGw = sG + wid * 16 * CH_;
        const int* sOw = sOff + wid * 16 * NC_;
        #pragma unroll 1
        for (int v = 0; v < 16; ++v) {
            int offs[NC_];
            #pragma unroll
            for (int c = 0; c < NC_; ++c) offs[c] = sOw[v * NC_ + c];
            #pragma unroll
            for (int k = 0; k < 8; ++k) {
                const int ch = lane + 32 * k;
                float s = 0.0f;
                #pragma unroll
                for (int c = 0; c < NC_; ++c)
                    if (offs[c] >= 0) s += img_feats[offs[c] + ch * HFWF_];
                sGw[v * CH_ + ch] = (__bf16)s;
            }
        }
    }

    __syncthreads();

    // ---- Phase 3: D(16x128) = A(16x256, bf16) x B(256x128, bf16) -------
    {
        __bf16* sGw = sG + wid * 16 * CH_;
        const int M     = lane & 15;   // A row / D column
        const int khalf = lane >> 4;

        // A-matrix regs, this code's contiguous k order (not the ISA table order):
        // half-wave khalf holds K = khalf*16 + j (matches B)
        v16bf a[8];
        #pragma unroll
        for (int kt = 0; kt < 8; ++kt) {
            const __bf16* src = sGw + M * CH_ + kt * 32 + khalf * 16;
            #pragma unroll
            for (int j = 0; j < 16; ++j) a[kt][j] = src[j];
        }

        // Per-wave sG region (8 KB) is now dead: reuse as f32 D bounce buffer.
        float* sD = (float*)sGw;       // [16][CIN_] f32 = 8192 B

        #pragma unroll
        for (int ot = 0; ot < 8; ++ot) {
            v8f acc = {};
            const int o = ot * 16 + M;           // output-channel column
            #pragma unroll
            for (int kt = 0; kt < 8; ++kt) {
                // B 32x16 layout: half-wave khalf holds K = kt*32 + khalf*16 + j
                v16bf bm;
                const __bf16* bs = sW + o * CH_ + kt * 32 + khalf * 16;
                #pragma unroll
                for (int j = 0; j < 16; ++j) bm[j] = bs[j];
                acc = __builtin_amdgcn_wmma_f32_16x16x32_bf16(
                        false, a[kt], false, bm, (short)0, acc, false, false);
            }
            // D layout: VGPR r -> row khalf*8 + r, column = M (bank-conflict-free)
            #pragma unroll
            for (int r = 0; r < 8; ++r)
                sD[(khalf * 8 + r) * CIN_ + o] = acc[r];
        }

        // Coalesced writeback: 16 rows x 128 f32 as float4 (b128 stores)
        const int tilebase = base + wid * 16;
        const float4* sD4 = (const float4*)sD;
        float4* dst = (float4*)(out_feats + (size_t)(N_ + tilebase) * CIN_);
        #pragma unroll
        for (int i = lane; i < 16 * (CIN_ / 4); i += 32) {
            if (FULL || (tilebase + (i >> 5)) < N_)
                dst[i] = sD4[i];
        }
    }
}

extern "C" void kernel_launch(void* const* d_in, const int* in_sizes, int n_in,
                              void* d_out, int out_size, void* d_ws, size_t ws_size,
                              hipStream_t stream) {
    const float* voxel_features = (const float*)d_in[0];
    const int*   voxel_coords   = (const int*)d_in[1];
    const float* img_feats      = (const float*)d_in[2];
    const float* intrins        = (const float*)d_in[3];
    const float* post_rots      = (const float*)d_in[4];
    const float* post_trans     = (const float*)d_in[5];
    const float* bda            = (const float*)d_in[6];
    const float* lidar2cam      = (const float*)d_in[7];
    const float* W              = (const float*)d_in[8];

    float* out_feats  = (float*)d_out;                                  // (2N, 128) f32
    int*   out_coords = (int*)((float*)d_out + (size_t)2 * N_ * CIN_);  // (2N, 4) i32 bits
    float* ws = (float*)d_ws;

    proj_prep_kernel<<<1, 64, 0, stream>>>(intrins, bda, lidar2cam, ws);

    const int nfull = N_ / VPB;                  // 1562 fully-populated tiles
    const int ntail = (N_ % VPB) ? 1 : 0;        // one partial tile (32 voxels)
    const size_t smem = 65536 + 32768 + 1536;    // W(bf16) + gsum(bf16) + offsets

    proj_view_kernel<true><<<nfull, TPB, smem, stream>>>(
        voxel_features, voxel_coords, img_feats, post_rots, post_trans,
        bda, W, ws, out_feats, out_coords, 0);
    if (ntail)
        proj_view_kernel<false><<<1, TPB, smem, stream>>>(
            voxel_features, voxel_coords, img_feats, post_rots, post_trans,
            bda, W, ws, out_feats, out_coords, nfull);
}
